// Residual_Moe_No_SAR_20083267076435
// MI455X (gfx1250) — matrix operand order repaired and checked by reading, not compiled, not run on hardware
//
#include <hip/hip_runtime.h>
#include <math.h>

typedef __attribute__((ext_vector_type(16))) __bf16       v16bf;
typedef __attribute__((ext_vector_type(8)))  float        v8f;
typedef __attribute__((ext_vector_type(8)))  unsigned int v8u;
typedef __attribute__((ext_vector_type(4)))  unsigned int u32x4;
typedef __attribute__((ext_vector_type(8)))  int          i32x8;
typedef __attribute__((ext_vector_type(4)))  int          i32x4;

#if __has_builtin(__builtin_amdgcn_tensor_load_to_lds)
#define HAVE_TDM 1
#endif

// ---------------------------------------------------------------------------
// helpers
// ---------------------------------------------------------------------------
__device__ __forceinline__ unsigned pack2(float a, float b) {
    unsigned short la = __builtin_bit_cast(unsigned short, (__bf16)a);
    unsigned short lb = __builtin_bit_cast(unsigned short, (__bf16)b);
    return (unsigned)la | ((unsigned)lb << 16);
}

__device__ __forceinline__ v8f wmma_bf16(v8u a, v8u b, v8f c) {
    return __builtin_amdgcn_wmma_f32_16x16x32_bf16(
        /*neg_a=*/false, __builtin_bit_cast(v16bf, a),
        /*neg_b=*/false, __builtin_bit_cast(v16bf, b),
        /*c_mod=*/(short)0, c, /*reuse_a=*/false, /*reuse_b=*/false);
}

#ifdef HAVE_TDM
// TDM 2D tile load: global (fp32, row-major, row_stride elems) -> LDS.
// D# per CDNA5 ISA ch.8: group0 {count, lds_addr, global_addr, type=2},
// group1 {data_size=4B, tensor_dim0/1, tile_dim0/1, tensor_dim0_stride}.
__device__ __forceinline__ void tdm_load_2d(unsigned lds_off, const void* gptr,
                                            unsigned tile_w, unsigned tile_h,
                                            unsigned tensor_w, unsigned tensor_h,
                                            unsigned row_stride)
{
    unsigned long long ga = (unsigned long long)gptr;
    u32x4 g0;
    g0[0] = 1u;                                             // count=1 (user)
    g0[1] = lds_off;                                        // lds_addr (bytes)
    g0[2] = (unsigned)(ga & 0xffffffffull);                 // global_addr lo
    g0[3] = (unsigned)((ga >> 32) & 0x1ffffffull) | (2u << 30); // hi | type=2
    i32x8 g1;
    g1[0] = (int)(2u << 16);                                // data_size = 4B
    g1[1] = (int)((tensor_w & 0xffffu) << 16);              // tensor_dim0 lo16
    g1[2] = (int)((tensor_w >> 16) | ((tensor_h & 0xffffu) << 16));
    g1[3] = (int)((tensor_h >> 16) | (tile_w << 16));       // tile_dim0
    g1[4] = (int)(tile_h & 0xffffu);                        // tile_dim1
    g1[5] = (int)row_stride;                                // dim0_stride lo32
    g1[6] = 0;
    g1[7] = 0;
    i32x4 z4 = {0, 0, 0, 0};
#if defined(__clang_major__) && (__clang_major__ >= 23)
    i32x8 z8 = {0, 0, 0, 0, 0, 0, 0, 0};
    __builtin_amdgcn_tensor_load_to_lds(g0, g1, z4, z4, z8, 0);
#else
    __builtin_amdgcn_tensor_load_to_lds(g0, g1, z4, z4, 0);
#endif
}
#endif

// ---------------------------------------------------------------------------
// Tiled GEMM:  C[M,N] = act(A[M,K] @ W[K,N] + bias)
// 256 threads = 8 waves laid out 4(M) x 2(N); block tile 64(M) x 128(N),
// K-step 32.  A tile staged fp32 by the Tensor Data Mover (wave 0 issues,
// s_wait_tensorcnt + barrier), B tile staged bf16 [n][k] by all threads.
// Per wave per K-step: 1 A fragment x 4 B fragments -> 4 WMMAs.
// ---------------------------------------------------------------------------
#define TM 64
#define TN 128
#define TK 32

__global__ __launch_bounds__(256) void gemm_bf16_wmma(
    const float* __restrict__ A, const float* __restrict__ W,
    const float* __restrict__ bias, float* __restrict__ C,
    int M, int N, int K, int relu, int permS, int permB)
{
    __shared__ __align__(16) float  sAf[TM * TK];   // fp32 A tile [m][k]
    __shared__ __bf16 sB[TN * TK];                  // bf16 B tile [n][k]

    const int tid  = threadIdx.x;
    const int lane = tid & 31;
    const int wid  = tid >> 5;
    const int wm   = wid >> 1;        // 0..3
    const int wn   = wid & 1;         // 0..1
    const int lrow = lane & 15;
    const int hi   = lane >> 4;       // 0 or 1
    const int m0   = blockIdx.y * TM;
    const int n0   = blockIdx.x * TN;

    v8f acc[4] = {};
    const unsigned int* pB = (const unsigned int*)sB;
    const float2* pA2 = (const float2*)sAf;
#ifdef HAVE_TDM
    const unsigned ldsA = (unsigned)(unsigned long long)sAf;
#endif

    for (int k0 = 0; k0 < K; k0 += TK) {
        // ---- stage A tile (64m x 32k fp32) via Tensor Data Mover ----
#ifdef HAVE_TDM
        if (wid == 0) {
            tdm_load_2d(ldsA, A + (size_t)m0 * K + k0,
                        TK, TM, (unsigned)K, (unsigned)M, (unsigned)K);
        }
#else
        for (int it = 0; it < 8; ++it) {            // fallback path
            int e = tid + it * 256;
            int r = e >> 5, kk = e & 31;
            sAf[e] = A[(size_t)(m0 + r) * K + (k0 + kk)];
        }
#endif
        // ---- stage B tile (32k x 128n fp32 -> bf16 [n][k]) ----
        for (int it = 0; it < 16; ++it) {
            int e  = tid + it * 256;
            int kk = e >> 7, n = e & 127;
            sB[n * TK + kk] = (__bf16)W[(size_t)(k0 + kk) * N + (n0 + n)];
        }
        if (k0 + TK < K)                // next W tile toward L2
            __builtin_prefetch(&W[(size_t)(k0 + TK + (tid >> 5)) * N + n0], 0, 1);
#ifdef HAVE_TDM
        if (wid == 0) __builtin_amdgcn_s_wait_tensorcnt(0);
#endif
        __syncthreads();

        // ---- A fragment: 16x32 bf16, ISA layout (lanes 0-15: K 0-7/16-23) --
        v8u au;
        {
            int ab = ((wm * 16 + lrow) * TK + hi * 8) >> 1;   // float2 index
            #pragma unroll
            for (int j = 0; j < 4; ++j) {
                float2 t = pA2[ab + j];
                float2 u = pA2[ab + 8 + j];
                au[j]     = pack2(t.x, t.y);     // K = kb+2j, +1
                au[j + 4] = pack2(u.x, u.y);     // K = 16+kb+2j, +1
            }
        }
        // ---- 4 B fragments (lanes 0-15: K 0-7/16-23; lanes 16-31: K 8-15/24-31) ----
        #pragma unroll
        for (int sub = 0; sub < 4; ++sub) {
            v8u bu;
            int bb = (wn * 64 + sub * 16 + lrow) * (TK / 2) + hi * 4;
            #pragma unroll
            for (int j = 0; j < 8; ++j) bu[j] = pB[bb + (j & 3) + ((j >> 2) << 3)];
            acc[sub] = wmma_bf16(au, bu, acc[sub]);
        }
        __syncthreads();
    }

    // ---- epilogue: C layout — VGPR r: lanes0-15 M=r, lanes16-31 M=r+8 ----
    const int rbase = hi * 8;
    #pragma unroll
    for (int sub = 0; sub < 4; ++sub) {
        int col = n0 + wn * 64 + sub * 16 + lrow;
        float bb = bias ? bias[col] : 0.f;
        #pragma unroll
        for (int r = 0; r < 8; ++r) {
            int m = m0 + wm * 16 + rbase + r;
            float v = acc[sub][r] + bb;
            if (relu) v = fmaxf(v, 0.f);
            size_t orow = permS ? ((size_t)(m % permB) * permS + (size_t)(m / permB))
                                : (size_t)m;
            C[orow * N + col] = v;
        }
    }
}

// ---------------------------------------------------------------------------
// Attention: one block per (b,h).  S=64, dh=64, NHEADS=16.
// scores = q k^T / 8  (WMMA)  -> softmax in LDS -> o = attn @ v (WMMA)
// qkv layout: [(s*B+b)*3D + {0,D,2D} + h*64 + d]
// ---------------------------------------------------------------------------
__global__ __launch_bounds__(256) void attn_kernel(
    const float* __restrict__ qkv, float* __restrict__ out,
    int S, int B, int D)
{
    __shared__ float  sc[64 * 64];
    __shared__ __bf16 at[64 * 64];

    const int bh = blockIdx.x;
    const int b = bh >> 4, h = bh & 15;
    const int tid = threadIdx.x, lane = tid & 31, wid = tid >> 5;
    const int lrow = lane & 15, hi = lane >> 4;
    const size_t RS = (size_t)B * 3 * D;                 // stride over s (or t)
    const float* qb = qkv + (size_t)b * 3 * D + h * 64;
    const float* kb = qb + D;
    const float* vb = qb + 2 * D;

    // ---- phase 1: scores (4x4 grid of 16x16 tiles, 2 tiles/wave) ----
    for (int tt = wid; tt < 16; tt += 8) {
        int ti = tt >> 2, tj = tt & 3;
        v8f c = {};
        #pragma unroll
        for (int d0 = 0; d0 < 64; d0 += 32) {
            v8u au, bu;
            int s  = ti * 16 + lrow;
            int ka = d0 + hi * 8;
            #pragma unroll
            for (int j = 0; j < 4; ++j) {
                int d = ka + 2 * j;
                au[j]     = pack2(qb[s * RS + d],      qb[s * RS + d + 1]);
                au[j + 4] = pack2(qb[s * RS + d + 16], qb[s * RS + d + 17]);
            }
            int t  = tj * 16 + lrow;
            int kd = d0 + hi * 8;
            #pragma unroll
            for (int j = 0; j < 8; ++j) {
                int d = kd + 2 * (j & 3) + ((j >> 2) << 4);
                bu[j] = pack2(kb[t * RS + d], kb[t * RS + d + 1]);
            }
            c = wmma_bf16(au, bu, c);
        }
        #pragma unroll
        for (int r = 0; r < 8; ++r) {
            int s = ti * 16 + hi * 8 + r;
            int t = tj * 16 + lrow;
            sc[s * 64 + t] = c[r] * 0.125f;              // 1/sqrt(dh)
        }
    }
    __syncthreads();

    // ---- softmax, one row per thread (rows 0..63) ----
    if (tid < 64) {
        float mx = -3.4e38f;
        for (int t = 0; t < 64; ++t) mx = fmaxf(mx, sc[tid * 64 + t]);
        float sum = 0.f;
        for (int t = 0; t < 64; ++t) {
            float e = __expf(sc[tid * 64 + t] - mx);
            sc[tid * 64 + t] = e; sum += e;
        }
        float inv = 1.f / sum;
        for (int t = 0; t < 64; ++t)
            at[tid * 64 + t] = (__bf16)(sc[tid * 64 + t] * inv);
    }
    __syncthreads();

    // ---- phase 2: o = attn @ v ----
    const unsigned int* pAt = (const unsigned int*)at;
    for (int tt = wid; tt < 16; tt += 8) {
        int ti = tt >> 2, tj = tt & 3;
        v8f c = {};
        #pragma unroll
        for (int t0 = 0; t0 < 64; t0 += 32) {
            v8u au, bu;
            int s = ti * 16 + lrow;
            int abase = (s * 64 + t0 + hi * 8) >> 1;
            #pragma unroll
            for (int j = 0; j < 4; ++j) {
                au[j]     = pAt[abase + j];
                au[j + 4] = pAt[abase + 8 + j];
            }
            int d  = tj * 16 + lrow;
            int kt = t0 + hi * 8;
            #pragma unroll
            for (int j = 0; j < 8; ++j) {
                int t = kt + 2 * (j & 3) + ((j >> 2) << 4);
                bu[j] = pack2(vb[t * RS + d], vb[(t + 1) * RS + d]);
            }
            c = wmma_bf16(au, bu, c);
        }
        #pragma unroll
        for (int r = 0; r < 8; ++r) {
            int s = ti * 16 + hi * 8 + r;
            int d = tj * 16 + lrow;
            out[((size_t)s * B + b) * D + h * 64 + d] = c[r];
        }
    }
}

// ---------------------------------------------------------------------------
// Fused residual + LayerNorm: out[m,:] = LN(x[m,:] + y[m,:]) * g + b, D=1024
// ---------------------------------------------------------------------------
__global__ __launch_bounds__(256) void resln_kernel(
    const float* __restrict__ x, const float* __restrict__ y,
    const float* __restrict__ g, const float* __restrict__ bta,
    float* __restrict__ out, int D)
{
    __shared__ float red[256];
    const size_t m = blockIdx.x;
    const float* xr = x + m * D;
    const float* yr = y + m * D;
    float v[4]; float s = 0.f, s2 = 0.f;
    #pragma unroll
    for (int i = 0; i < 4; ++i) {
        int d = threadIdx.x + i * 256;
        float t = xr[d] + yr[d];
        v[i] = t; s += t; s2 += t * t;
    }
    red[threadIdx.x] = s; __syncthreads();
    for (int o = 128; o > 0; o >>= 1) {
        if (threadIdx.x < o) red[threadIdx.x] += red[threadIdx.x + o];
        __syncthreads();
    }
    float mean = red[0] / D; __syncthreads();
    red[threadIdx.x] = s2; __syncthreads();
    for (int o = 128; o > 0; o >>= 1) {
        if (threadIdx.x < o) red[threadIdx.x] += red[threadIdx.x + o];
        __syncthreads();
    }
    float var = red[0] / D - mean * mean;
    float inv = rsqrtf(var + 1e-5f);
    #pragma unroll
    for (int i = 0; i < 4; ++i) {
        int d = threadIdx.x + i * 256;
        out[m * D + d] = (v[i] - mean) * inv * g[d] + bta[d];
    }
}

// ---------------------------------------------------------------------------
// g[b, 0:2D] = concat(cat[0,b,:], cat[1,b,:])
// ---------------------------------------------------------------------------
__global__ void build_g(const float* __restrict__ cat, float* __restrict__ g,
                        int B, int D)
{
    int i = blockIdx.x * blockDim.x + threadIdx.x;
    int tot = B * 2 * D;
    if (i >= tot) return;
    int b = i / (2 * D);
    int k = i % (2 * D);
    int s = k / D, d = k % D;
    g[i] = cat[((size_t)s * B + b) * D + d];
}

// ---------------------------------------------------------------------------
// Gate finish: logits = hidden @ W2 + b2 (N=5), softmax, argmax, cascaded
// ---------------------------------------------------------------------------
__global__ __launch_bounds__(256) void gate_finish(
    const float* __restrict__ hidden, const float* __restrict__ W2,
    const float* __restrict__ b2, float* __restrict__ gcoef, int B)
{
    int b = blockIdx.x * blockDim.x + threadIdx.x;
    if (b >= B) return;
    float lg[5];
    #pragma unroll
    for (int j = 0; j < 5; ++j) lg[j] = b2[j];
    for (int i = 0; i < 256; ++i) {
        float hv = hidden[b * 256 + i];
        #pragma unroll
        for (int j = 0; j < 5; ++j) lg[j] += hv * W2[i * 5 + j];
    }
    float mx = lg[0]; int ks = 0;
    #pragma unroll
    for (int j = 1; j < 5; ++j) if (lg[j] > mx) { mx = lg[j]; ks = j; }
    float p[5], sum = 0.f;
    #pragma unroll
    for (int j = 0; j < 5; ++j) { p[j] = __expf(lg[j] - mx); sum += p[j]; }
    float inv = 1.f / sum;
    #pragma unroll
    for (int j = 0; j < 5; ++j) p[j] *= inv;
    float ww[4];
    ww[3] = p[4];
    ww[2] = p[3] + ww[3];
    ww[1] = p[2] + ww[2];
    ww[0] = p[1] + ww[1];
    #pragma unroll
    for (int e = 0; e < 4; ++e)
        gcoef[e * B + b] = (ks >= e + 1) ? ww[e] : 0.f;
}

// ---------------------------------------------------------------------------
// acc[m,d] += gcoef[e, m % B] * proj[m,d]
// ---------------------------------------------------------------------------
__global__ void accum_kernel(float* __restrict__ acc,
                             const float* __restrict__ proj,
                             const float* __restrict__ gcoef,
                             int e, int B, int D, size_t total)
{
    size_t i = (size_t)blockIdx.x * 1024 + threadIdx.x;
    #pragma unroll
    for (int j = 0; j < 4; ++j) {
        size_t idx = i + (size_t)j * 256;
        if (idx < total) {
            size_t m = idx / (size_t)D;
            int b = (int)(m % (size_t)B);
            acc[idx] += gcoef[e * B + b] * proj[idx];
        }
    }
}

__global__ void write_scalar(float* p) { *p = 0.0f; }

// ---------------------------------------------------------------------------
// host orchestration
// ---------------------------------------------------------------------------
extern "C" void kernel_launch(void* const* d_in, const int* in_sizes, int n_in,
                              void* d_out, int out_size, void* d_ws, size_t ws_size,
                              hipStream_t stream)
{
    (void)in_sizes; (void)n_in; (void)out_size; (void)ws_size;
    constexpr int S = 64, Bb = 256, D = 1024, F = 4096, E = 4, OUTD = 512;
    const size_t M = (size_t)S * Bb;

    auto in = [&](int i) { return (const float*)d_in[i]; };
    const float* cat = in(0);
    const float* gW1 = in(1); const float* gb1 = in(2);
    const float* gW2 = in(3); const float* gb2 = in(4);
    const float* P[2][12];
    for (int j = 0; j < 12; ++j) { P[0][j] = in(5 + j); P[1][j] = in(17 + j); }
    const float* zeroW = in(29); const float* zerob = in(30);
    const float* outW  = in(31); const float* outb  = in(32);
    float* out = (float*)d_out;

    float* ws = (float*)d_ws;
    size_t off = 0;
    float* g    = ws + off; off += (size_t)Bb * 2 * D;   // [B, 2D]
    float* hid  = ws + off; off += (size_t)Bb * 256;     // [B, 256]
    float* gco  = ws + off; off += (size_t)E * Bb;       // [E, B]
    float* big  = ws + off; off += M * F;                // qkv [M,3D] / ffn [M,F]
    float* att  = ws + off; off += M * D;
    float* tmp  = ws + off; off += M * D;                // also expert proj
    float* x1   = ws + off; off += M * D;
    float* x2   = ws + off; off += M * D;
    float* acc  = ws + off; off += M * D;

    auto GEMM = [&](const float* A, const float* W, const float* bias, float* C,
                    int Mm, int Nn, int Kk, int relu, int permS, int permB) {
        dim3 grid(Nn / TN, Mm / TM);
        gemm_bf16_wmma<<<grid, 256, 0, stream>>>(A, W, bias, C, Mm, Nn, Kk,
                                                 relu, permS, permB);
    };

    // ---- gate ----
    build_g<<<(Bb * 2 * D + 255) / 256, 256, 0, stream>>>(cat, g, Bb, D);
    GEMM(g, gW1, gb1, hid, Bb, 256, 2 * D, /*relu=*/1, 0, 1);
    gate_finish<<<1, 256, 0, stream>>>(hid, gW2, gb2, gco, Bb);

    // ---- base + experts ----
    const size_t psz[12] = {(size_t)D * 3 * D, 3 * (size_t)D, (size_t)D * D,
                            (size_t)D, (size_t)D, (size_t)D,
                            (size_t)D * F, (size_t)F, (size_t)F * D,
                            (size_t)D, (size_t)D, (size_t)D};
    for (int layer = 0; layer < 1 + E; ++layer) {
        const float* p[12];
        if (layer == 0) {
            for (int j = 0; j < 12; ++j) p[j] = P[0][j];
        } else {
            int e = layer - 1;
            for (int j = 0; j < 12; ++j) p[j] = P[1][j] + (size_t)e * psz[j];
        }
        const float *Wqkv = p[0], *bqkv = p[1], *Wo = p[2], *bo = p[3];
        const float *l1g = p[4], *l1b = p[5];
        const float *W1 = p[6], *b1 = p[7], *W2 = p[8], *b2 = p[9];
        const float *l2g = p[10], *l2b = p[11];

        GEMM(cat, Wqkv, bqkv, big, (int)M, 3 * D, D, 0, 0, 1);        // qkv
        attn_kernel<<<Bb * 16, 256, 0, stream>>>(big, att, S, Bb, D); // attn
        GEMM(att, Wo, bo, tmp, (int)M, D, D, 0, 0, 1);                // o @ Wo
        resln_kernel<<<(int)M, 256, 0, stream>>>(cat, tmp, l1g, l1b, x1, D);
        GEMM(x1, W1, b1, big, (int)M, F, D, /*relu=*/1, 0, 1);        // FFN up
        GEMM(big, W2, b2, tmp, (int)M, D, F, 0, 0, 1);                // FFN down
        resln_kernel<<<(int)M, 256, 0, stream>>>(x1, tmp, l2g, l2b, x2, D);

        if (layer == 0) {
            hipMemcpyAsync(acc, x2, M * D * sizeof(float),
                           hipMemcpyDeviceToDevice, stream);          // emb
        } else {
            int e = layer - 1;
            GEMM(x2, zeroW + (size_t)e * D * D, zerob + (size_t)e * D,
                 tmp, (int)M, D, D, 0, 0, 1);                         // zero proj
            accum_kernel<<<(unsigned)(M * D / 1024), 256, 0, stream>>>(
                acc, tmp, gco, e, Bb, D, M * D);
        }
    }

    // ---- final projection, output permuted [S,B]->[B,S] rows ----
    GEMM(acc, outW, outb, out, (int)M, OUTD, D, 0, /*permS=*/S, /*permB=*/Bb);
    write_scalar<<<1, 1, 0, stream>>>(out + M * OUTD);
}
